// TemporalAttention_9174050144615
// MI455X (gfx1250) — matrix operand order repaired and checked by reading, not compiled, not run on hardware
//
#include <hip/hip_runtime.h>

typedef __attribute__((ext_vector_type(16))) __bf16 v16bf;
typedef __attribute__((ext_vector_type(8)))  __bf16 v8bf;
typedef __attribute__((ext_vector_type(8)))  float  v8f;

constexpr int kHeads = 8;
constexpr int kDH    = 32;     // head dim
constexpr int kHid   = 256;    // heads * dh
constexpr int kF     = 32;     // tokens (frames)
constexpr int kC     = 256;    // channels
constexpr int kPos   = 1024;   // h*w spatial positions per batch
constexpr float kScale = 0.17677669529663689f;  // 32^-0.5

// LDS row strides (bf16 elements); multiples of 8 (=16B) for b128 loads
constexpr int SX = 264;        // x tile / attn-out tile rows (528B)
constexpr int SQ = 40;         // q/k/vT rows (80B)

// packed-weight fragment geometry: one 32x16 (KxN) B fragment = 512 bf16 (1KB):
//   [0,256)  : elems 0..7  for lanes 0..31 (8 bf16 each, contiguous)
//   [256,512): elems 8..15 for lanes 0..31
constexpr int kFragElems  = 512;
constexpr int kQkvFrags   = (3 * kHid / 16) * 8;   // 48 ntiles * 8 ksteps = 384
constexpr int kOutFrags   = (kHid / 16) * 8;       // 128
constexpr size_t kPackBytes = (size_t)(kQkvFrags + kOutFrags) * kFragElems * 2; // 512KB

__device__ __forceinline__ v8f wmma_bf16(v16bf a, v16bf b, v8f c) {
    return __builtin_amdgcn_wmma_f32_16x16x32_bf16(false, a, false, b, (short)0, c, false, false);
}

__device__ __forceinline__ v8f vzero8() {
    v8f z;
#pragma unroll
    for (int j = 0; j < 8; ++j) z[j] = 0.0f;
    return z;
}

// A fragment (16x32, 16-bit): lanes 0-15 -> M=lane, elems K0..15;
// lanes 16-31 -> M=lane-16, elems K16..31. Source row-major in LDS.
__device__ __forceinline__ v16bf lds_A(const __bf16* base, int stride, int lane,
                                       int mtile, int kcol) {
    const int m  = (lane & 15) + mtile * 16;
    const int kb = (lane & 16) ? 16 : 0;
    const __bf16* r = base + m * stride + kcol + kb;
    v8bf lo = *(const v8bf*)(r);
    v8bf hi = *(const v8bf*)(r + 8);
    v16bf a;
#pragma unroll
    for (int j = 0; j < 8; ++j) { a[j] = lo[j]; a[j + 8] = hi[j]; }
    return a;
}

// B fragment (32x16 KxN): lane holds 16 consecutive K of column n = lane%16,
// K-offset 0 (lanes 0-15) / 16 (lanes 16-31). Source: column contiguous in LDS row n.
__device__ __forceinline__ v16bf lds_B(const __bf16* base, int stride, int lane,
                                       int ntile, int kcol) {
    const int n  = (lane & 15) + ntile * 16;
    const int kb = (lane & 16) ? 16 : 0;
    const __bf16* r = base + n * stride + kcol + kb;
    v8bf lo = *(const v8bf*)(r);
    v8bf hi = *(const v8bf*)(r + 8);
    v16bf b;
#pragma unroll
    for (int j = 0; j < 8; ++j) { b[j] = lo[j]; b[j + 8] = hi[j]; }
    return b;
}

// B fragment from pre-packed bf16 weights: two coalesced b128 loads per lane.
__device__ __forceinline__ v16bf pk_B(const __bf16* __restrict__ pk, int frag, int lane) {
    const __bf16* base = pk + (long)frag * kFragElems;
    v8bf lo = *(const v8bf*)(base + lane * 8);
    v8bf hi = *(const v8bf*)(base + 256 + lane * 8);
    v16bf b;
#pragma unroll
    for (int j = 0; j < 8; ++j) { b[j] = lo[j]; b[j + 8] = hi[j]; }
    return b;
}

// Fallback: B fragment streamed from row-major f32 W[n][k] (GEMM B = W^T).
__device__ __forceinline__ v16bf glb_B(const float* __restrict__ w, int rowbase,
                                       int ldw, int lane, int ntile, int kcol) {
    const int n  = rowbase + ntile * 16 + (lane & 15);
    const int k0 = kcol + ((lane & 16) ? 16 : 0);
    const float4* p = (const float4*)(w + (long)n * ldw + k0);
    v16bf b;
#pragma unroll
    for (int q = 0; q < 4; ++q) {
        float4 f = p[q];
        b[q * 4 + 0] = (__bf16)f.x;
        b[q * 4 + 1] = (__bf16)f.y;
        b[q * 4 + 2] = (__bf16)f.z;
        b[q * 4 + 3] = (__bf16)f.w;
    }
    return b;
}

// ---- pre-pack both weight matrices into B-fragment-ordered bf16 (one wave/frag) ----
__global__ __launch_bounds__(32) void prepack_weights(
    const float* __restrict__ w_qkv, const float* __restrict__ w_out,
    __bf16* __restrict__ pk) {
    const int f    = blockIdx.x;
    const int lane = threadIdx.x;
    const float* w;
    int nt, kk;
    __bf16* dst;
    if (f < kQkvFrags) { w = w_qkv; nt = f >> 3; kk = f & 7; dst = pk + (long)f * kFragElems; }
    else { const int g = f - kQkvFrags; w = w_out; nt = g >> 3; kk = g & 7;
           dst = pk + (long)(kQkvFrags + g) * kFragElems; }
    const int n  = nt * 16 + (lane & 15);
    const int k0 = kk * 32 + ((lane & 16) ? 16 : 0);
    const float4* p = (const float4*)(w + (long)n * 256 + k0);
    v8bf lo, hi;
#pragma unroll
    for (int q = 0; q < 2; ++q) {
        float4 a = p[q], b = p[q + 2];
        lo[q * 4 + 0] = (__bf16)a.x; lo[q * 4 + 1] = (__bf16)a.y;
        lo[q * 4 + 2] = (__bf16)a.z; lo[q * 4 + 3] = (__bf16)a.w;
        hi[q * 4 + 0] = (__bf16)b.x; hi[q * 4 + 1] = (__bf16)b.y;
        hi[q * 4 + 2] = (__bf16)b.z; hi[q * 4 + 3] = (__bf16)b.w;
    }
    *(v8bf*)(dst + lane * 8)       = lo;
    *(v8bf*)(dst + 256 + lane * 8) = hi;
}

template <bool PK>
__global__ __launch_bounds__(256) void temporal_attn_fused(
    const float* __restrict__ x, const float* __restrict__ w_qkv,
    const float* __restrict__ w_out, const float* __restrict__ b_out,
    const float* __restrict__ rel, const __bf16* __restrict__ pk,
    float* __restrict__ out) {

    // 159.2 KB total -> 2 workgroups per 320KB WGP
    __shared__ __bf16 s_xo[2][kF][SX];            // x tile (t,c); later attn-out (t,hid)
    __shared__ __bf16 s_q[2][kHeads][kF][SQ];     // q (t,d); reused for probs
    __shared__ __bf16 s_k[2][kHeads][kF][SQ];     // k (t,d)
    __shared__ __bf16 s_vT[2][kHeads][kDH][SQ];   // v^T (d,t)
    __shared__ float  s_rel[199][kHeads];

    const int tid  = threadIdx.x;
    const int h    = tid >> 5;                    // wave id == head id
    const int lane = tid & 31;
    const int gp0  = blockIdx.x * 2;              // first of 2 adjacent positions
    const int b    = gp0 >> 10;
    const int p0   = gp0 & 1023;                  // even
    const long xbase = (long)b * (kC * kF * kPos) + p0;  // + (ch*kF + t)*kPos (+0/1)

    const int nloc = lane & 15;
    const int mb   = (lane & 16) ? 8 : 0;

    // ---- Phase 0: stage x for both positions (float2: adjacent in memory) ----
    for (int e = tid; e < kF * kC; e += 256) {    // e = ch*32 + t
        const int ch = e >> 5, t = e & 31;
        float2 v = *(const float2*)(x + xbase + (long)e * kPos);
        s_xo[0][t][ch] = (__bf16)v.x;
        s_xo[1][t][ch] = (__bf16)v.y;
    }
    for (int e = tid; e < 199 * kHeads; e += 256)
        s_rel[e >> 3][e & 7] = rel[e];
    __syncthreads();

    // ---- Phase 1: QKV GEMM; B fragments shared across both positions ----
#pragma unroll
    for (int mat = 0; mat < 3; ++mat) {
        v8f acc[2][2][2];                          // [pos][mt][nt]
#pragma unroll
        for (int ps = 0; ps < 2; ++ps)
#pragma unroll
            for (int mt = 0; mt < 2; ++mt)
#pragma unroll
                for (int nt = 0; nt < 2; ++nt) acc[ps][mt][nt] = vzero8();

        const int nt16 = mat * 16 + h * 2;         // 16-row tile index into w_qkv
#pragma unroll
        for (int kk = 0; kk < 8; ++kk) {
            const int kcol = kk * 32;
            v16bf b0, b1;
            if (PK) {
                b0 = pk_B(pk, (nt16 + 0) * 8 + kk, lane);
                b1 = pk_B(pk, (nt16 + 1) * 8 + kk, lane);
                if (kk < 7)  // hint the next 2KB of the packed stream into cache
                    __builtin_prefetch(pk + (long)((nt16 + 0) * 8 + kk + 1) * kFragElems, 0, 1);
            } else {
                b0 = glb_B(w_qkv, mat * kHid + h * kDH, kC, lane, 0, kcol);
                b1 = glb_B(w_qkv, mat * kHid + h * kDH, kC, lane, 1, kcol);
            }
#pragma unroll
            for (int ps = 0; ps < 2; ++ps) {
                v16bf a0 = lds_A(&s_xo[ps][0][0], SX, lane, 0, kcol);
                v16bf a1 = lds_A(&s_xo[ps][0][0], SX, lane, 1, kcol);
                acc[ps][0][0] = wmma_bf16(a0, b0, acc[ps][0][0]);
                acc[ps][0][1] = wmma_bf16(a0, b1, acc[ps][0][1]);
                acc[ps][1][0] = wmma_bf16(a1, b0, acc[ps][1][0]);
                acc[ps][1][1] = wmma_bf16(a1, b1, acc[ps][1][1]);
            }
        }
#pragma unroll
        for (int ps = 0; ps < 2; ++ps)
#pragma unroll
            for (int mt = 0; mt < 2; ++mt)
#pragma unroll
                for (int nt = 0; nt < 2; ++nt)
#pragma unroll
                    for (int r = 0; r < 8; ++r) {
                        const int M = mt * 16 + mb + r;
                        const int N = nt * 16 + nloc;
                        const float v = acc[ps][mt][nt][r];
                        if (mat == 0)      s_q[ps][h][M][N]  = (__bf16)(v * kScale);
                        else if (mat == 1) s_k[ps][h][M][N]  = (__bf16)v;
                        else               s_vT[ps][h][N][M] = (__bf16)v;
                    }
    }
    __syncthreads();

    // ---- Phase 2: scores + register softmax (shuffle reductions), probs -> s_q ----
#pragma unroll
    for (int ps = 0; ps < 2; ++ps) {
        v16bf a0 = lds_A(&s_q[ps][h][0][0], SQ, lane, 0, 0);
        v16bf a1 = lds_A(&s_q[ps][h][0][0], SQ, lane, 1, 0);
        v16bf b0 = lds_B(&s_k[ps][h][0][0], SQ, lane, 0, 0);
        v16bf b1 = lds_B(&s_k[ps][h][0][0], SQ, lane, 1, 0);
        v8f s00 = wmma_bf16(a0, b0, vzero8());
        v8f s01 = wmma_bf16(a0, b1, vzero8());
        v8f s10 = wmma_bf16(a1, b0, vzero8());
        v8f s11 = wmma_bf16(a1, b1, vzero8());
        // D layout: row i = mt*16 + mb + r lives in one 16-lane half; its 32 j-values
        // are 2 per lane (j = nloc, nloc+16). Reduce with xor-shuffles (masks < 16).
#pragma unroll
        for (int mt = 0; mt < 2; ++mt) {
            v8f sa = (mt == 0) ? s00 : s10;
            v8f sb = (mt == 0) ? s01 : s11;
#pragma unroll
            for (int r = 0; r < 8; ++r) {
                const int i = mt * 16 + mb + r;
                float a  = sa[r] + s_rel[nloc - i + 99][h];
                float bv = sb[r] + s_rel[nloc + 16 - i + 99][h];
                float mx = fmaxf(a, bv);
#pragma unroll
                for (int m = 1; m < 16; m <<= 1) mx = fmaxf(mx, __shfl_xor(mx, m, 32));
                const float ea = __expf(a - mx), eb = __expf(bv - mx);
                float sum = ea + eb;
#pragma unroll
                for (int m = 1; m < 16; m <<= 1) sum += __shfl_xor(sum, m, 32);
                const float inv = 1.0f / sum;
                s_q[ps][h][i][nloc]      = (__bf16)(ea * inv);
                s_q[ps][h][i][nloc + 16] = (__bf16)(eb * inv);
            }
        }
    }
    __syncthreads();

    // ---- Phase 3: out_h = probs @ v; write into s_xo (x tile is dead) ----
#pragma unroll
    for (int ps = 0; ps < 2; ++ps) {
        v16bf a0 = lds_A(&s_q[ps][h][0][0],  SQ, lane, 0, 0);
        v16bf a1 = lds_A(&s_q[ps][h][0][0],  SQ, lane, 1, 0);
        v16bf b0 = lds_B(&s_vT[ps][h][0][0], SQ, lane, 0, 0);
        v16bf b1 = lds_B(&s_vT[ps][h][0][0], SQ, lane, 1, 0);
        v8f o00 = wmma_bf16(a0, b0, vzero8());
        v8f o01 = wmma_bf16(a0, b1, vzero8());
        v8f o10 = wmma_bf16(a1, b0, vzero8());
        v8f o11 = wmma_bf16(a1, b1, vzero8());
#pragma unroll
        for (int r = 0; r < 8; ++r) {
            s_xo[ps][ 0 + mb + r][h * kDH +  0 + nloc] = (__bf16)o00[r];
            s_xo[ps][ 0 + mb + r][h * kDH + 16 + nloc] = (__bf16)o01[r];
            s_xo[ps][16 + mb + r][h * kDH +  0 + nloc] = (__bf16)o10[r];
            s_xo[ps][16 + mb + r][h * kDH + 16 + nloc] = (__bf16)o11[r];
        }
    }
    __syncthreads();

    // ---- Phase 4: out-proj slice (cout in [h*32,h*32+32)) + bias + residual ----
    {
        v8f f[2][2][2];                            // [pos][mt][nt]
#pragma unroll
        for (int ps = 0; ps < 2; ++ps)
#pragma unroll
            for (int mt = 0; mt < 2; ++mt)
#pragma unroll
                for (int nt = 0; nt < 2; ++nt) f[ps][mt][nt] = vzero8();

#pragma unroll
        for (int kk = 0; kk < 8; ++kk) {
            const int kcol = kk * 32;
            v16bf b0, b1;
            if (PK) {
                b0 = pk_B(pk, kQkvFrags + (h * 2 + 0) * 8 + kk, lane);
                b1 = pk_B(pk, kQkvFrags + (h * 2 + 1) * 8 + kk, lane);
            } else {
                b0 = glb_B(w_out, h * kDH, kHid, lane, 0, kcol);
                b1 = glb_B(w_out, h * kDH, kHid, lane, 1, kcol);
            }
#pragma unroll
            for (int ps = 0; ps < 2; ++ps) {
                v16bf a0 = lds_A(&s_xo[ps][0][0], SX, lane, 0, kcol);
                v16bf a1 = lds_A(&s_xo[ps][0][0], SX, lane, 1, kcol);
                f[ps][0][0] = wmma_bf16(a0, b0, f[ps][0][0]);
                f[ps][0][1] = wmma_bf16(a0, b1, f[ps][0][1]);
                f[ps][1][0] = wmma_bf16(a1, b0, f[ps][1][0]);
                f[ps][1][1] = wmma_bf16(a1, b1, f[ps][1][1]);
            }
        }
#pragma unroll
        for (int mt = 0; mt < 2; ++mt)
#pragma unroll
            for (int nt = 0; nt < 2; ++nt) {
                const int cout = h * kDH + nt * 16 + nloc;
                const float bq = b_out[cout];
#pragma unroll
                for (int r = 0; r < 8; ++r) {
                    const int t = mt * 16 + mb + r;
                    const long idx = xbase + (long)(cout * kF + t) * kPos;
                    float2 res = *(const float2*)(x + idx);   // residual: L2-hot
                    float2 o2;
                    o2.x = f[0][mt][nt][r] + bq + res.x;
                    o2.y = f[1][mt][nt][r] + bq + res.y;
                    *(float2*)(out + idx) = o2;               // 8B coalesced pair store
                }
            }
    }
}

extern "C" void kernel_launch(void* const* d_in, const int* in_sizes, int n_in,
                              void* d_out, int out_size, void* d_ws, size_t ws_size,
                              hipStream_t stream) {
    const float* x     = (const float*)d_in[0];
    const float* w_qkv = (const float*)d_in[1];
    const float* w_out = (const float*)d_in[2];
    const float* b_out = (const float*)d_in[3];
    const float* rel   = (const float*)d_in[4];
    float* out = (float*)d_out;

    if (ws_size >= kPackBytes) {
        __bf16* pk = (__bf16*)d_ws;
        prepack_weights<<<kQkvFrags + kOutFrags, 32, 0, stream>>>(w_qkv, w_out, pk);
        temporal_attn_fused<true><<<1024, 256, 0, stream>>>(x, w_qkv, w_out, b_out, rel,
                                                            pk, out);
    } else {
        temporal_attn_fused<false><<<1024, 256, 0, stream>>>(x, w_qkv, w_out, b_out, rel,
                                                             nullptr, out);
    }
}
